// Transformer_blk_51015621542505
// MI455X (gfx1250) — matrix operand order repaired and checked by reading, not compiled, not run on hardware
//
#include <hip/hip_runtime.h>
#include <hip/hip_bf16.h>

#define TB 256
#define D_EMB 1024
#define N_HEADS 16
#define HEAD_DIM 64
#define SEQ 2048
#define BATCH 4
#define FFDIM 4096

typedef __attribute__((ext_vector_type(16))) __bf16 v16bf;
typedef __attribute__((ext_vector_type(8)))  float  v8f;
typedef unsigned int u32x4 __attribute__((ext_vector_type(4)));
typedef int          i32x4 __attribute__((ext_vector_type(4)));
typedef int          i32x8 __attribute__((ext_vector_type(8)));

union AF {
  v16bf v;
  unsigned int w[8];
  unsigned short u[16];
};

__device__ __forceinline__ unsigned short f2bf(float f) {
  unsigned int u = __float_as_uint(f);
  unsigned int r = u + 0x7FFFu + ((u >> 16) & 1u);
  return (unsigned short)(r >> 16);
}

// K index of pair p (0..7) in the 16-bit 16x32 A-fragment (contiguous: k = 16*hi + 2p)
__device__ __forceinline__ int akp(int p, int hi) {
  int k = 2 * p;
  return k + (hi ? 16 : 0);
}

__device__ __forceinline__ v8f wmma_bf16(v16bf a, v16bf b, v8f c) {
  return __builtin_amdgcn_wmma_f32_16x16x32_bf16(
      false, a, false, b, (short)0, c, false, false);
}

__device__ __forceinline__ float red16_max(float v) {
  v = fmaxf(v, __shfl_xor(v, 1, 32));
  v = fmaxf(v, __shfl_xor(v, 2, 32));
  v = fmaxf(v, __shfl_xor(v, 4, 32));
  v = fmaxf(v, __shfl_xor(v, 8, 32));
  return v;
}
__device__ __forceinline__ float red16_sum(float v) {
  v += __shfl_xor(v, 1, 32);
  v += __shfl_xor(v, 2, 32);
  v += __shfl_xor(v, 4, 32);
  v += __shfl_xor(v, 8, 32);
  return v;
}

// low 32 bits of a generic pointer to __shared__ == LDS byte offset
__device__ __forceinline__ unsigned lds_off(const void* p) {
  return (unsigned)(unsigned long long)p;
}

// async global->LDS copy of 16B, per-lane (tracked by ASYNCcnt)
__device__ __forceinline__ void async_load_b128(unsigned dst_lds,
                                                const void* gaddr) {
  asm volatile("global_load_async_to_lds_b128 %0, %1, off"
               :
               : "v"(dst_lds), "v"(gaddr)
               : "memory");
}

// TDM: 2D tile (elements of 2 bytes) global -> LDS, with LDS row padding.
// tile_w elements per row, tile_h rows, row stride = stride_elems.
// pad_int_code: dwords-per-pad-interval code (0=2,1=4,2=8,3=16,4=32,5=64,...)
// pad_amt_code: pad dwords minus 1.
__device__ __forceinline__ void tdm_load_2d_bf16(unsigned dst_lds,
                                                 const void* gptr,
                                                 unsigned tile_w,
                                                 unsigned tile_h,
                                                 unsigned stride_elems,
                                                 unsigned pad_int_code,
                                                 unsigned pad_amt_code) {
  unsigned long long ga = (unsigned long long)gptr;
  u32x4 g0;
  g0.x = 0x1u;                                   // count=1, no gather
  g0.y = dst_lds;                                // lds_addr
  g0.z = (unsigned)(ga & 0xFFFFFFFFu);           // global_addr lo
  g0.w = (unsigned)((ga >> 32) & 0x01FFFFFFu) | (2u << 30);  // hi + type=2
  i32x8 g1;
  g1[0] = (int)((1u << 16) |                     // data_size = 2 bytes
                (1u << 20) |                     // pad_enable
                (pad_int_code << 22) | (pad_amt_code << 25));
  g1[1] = (int)(0xFFFFu << 16);                  // tensor_dim0 lo16 (huge)
  g1[2] = (int)((0xFFFFu << 16) | 0x7FFFu);      // td0 hi16 | td1 lo16
  g1[3] = (int)((tile_w << 16) | 0x7FFFu);       // td1 hi16 | tile_dim0
  g1[4] = (int)(tile_h & 0xFFFFu);               // tile_dim1 (tile_dim2=0)
  g1[5] = (int)stride_elems;                     // tensor_dim0_stride lo32
  g1[6] = 0;                                     // stride hi16 | td1_stride lo
  g1[7] = 0;
  i32x4 gz = {0, 0, 0, 0};
#if __clang_major__ >= 23
  i32x8 gz8 = {0, 0, 0, 0, 0, 0, 0, 0};
  __builtin_amdgcn_tensor_load_to_lds(g0, g1, gz, gz, gz8, 0);
#else
  __builtin_amdgcn_tensor_load_to_lds(g0, g1, gz, gz, 0);
#endif
}

// ---------------- fp32 -> bf16 weight conversion ----------------
__global__ void cvt_f32_bf16(const float* __restrict__ in,
                             unsigned short* __restrict__ out, int n) {
  int i = blockIdx.x * blockDim.x + threadIdx.x;
  int stride = gridDim.x * blockDim.x;
  for (; i < n; i += stride) out[i] = f2bf(in[i]);
}

// ---------------- LayerNorm -> bf16 ----------------
__global__ __launch_bounds__(TB) void layernorm_bf16(
    const float* __restrict__ x, const float* __restrict__ g,
    const float* __restrict__ sh, unsigned short* __restrict__ out) {
  __shared__ float red[TB];
  const int row = blockIdx.x;
  const int tid = threadIdx.x;
  const float* xr = x + (size_t)row * D_EMB;

  float s = 0.f;
  for (int c = tid; c < D_EMB; c += TB) s += xr[c];
  red[tid] = s;
  __syncthreads();
  for (int o = TB / 2; o > 0; o >>= 1) {
    if (tid < o) red[tid] += red[tid + o];
    __syncthreads();
  }
  const float mean = red[0] / (float)D_EMB;
  __syncthreads();

  float v = 0.f;
  for (int c = tid; c < D_EMB; c += TB) {
    float d = xr[c] - mean;
    v += d * d;
  }
  red[tid] = v;
  __syncthreads();
  for (int o = TB / 2; o > 0; o >>= 1) {
    if (tid < o) red[tid] += red[tid + o];
    __syncthreads();
  }
  const float inv = rsqrtf(red[0] / (float)D_EMB + 1e-15f);

  unsigned short* orow = out + (size_t)row * D_EMB;
  for (int c = tid; c < D_EMB; c += TB)
    orow[c] = f2bf(g[c] * (xr[c] - mean) * inv + sh[c]);
}

// ---------------- bf16 WMMA GEMM with TDM-staged, double-buffered tiles ----
// C[M,N] = A[M,K] * W[K,N]; block tile 128x128, BK=32, 8 waves (4x2),
// wave tile 32x64 (2x4 wmma accs). Wave 0 drives the Tensor Data Mover.
template <bool BIAS, bool RESID, bool GELU_, bool OUTBF>
__global__ __launch_bounds__(TB) void gemm_wmma_bf16(
    const unsigned short* __restrict__ A, const unsigned short* __restrict__ W,
    const float* __restrict__ bias, const float* __restrict__ resid,
    float* __restrict__ outF, unsigned short* __restrict__ outB,
    int M, int N, int K) {
  // LDS row strides include the padding inserted by the TDM descriptor:
  // A rows: 32 el (16 dw) + 4 dw pad  -> stride 40 elements
  // B rows: 128 el (64 dw) + 4 dw pad -> stride 136 elements
  __shared__ unsigned short As[2][128][40];  // [buf][m][k]
  __shared__ unsigned short Bs[2][32][136];  // [buf][k][n]

  const int tid = threadIdx.x;
  const int wid = tid >> 5, lane = tid & 31;
  const int hi = lane >> 4, l16 = lane & 15;
  const int wM = (wid >> 1) * 32;
  const int wN = (wid & 1) * 64;
  const int bM = blockIdx.y * 128;
  const int bN = blockIdx.x * 128;

  v8f acc[2][4];
#pragma unroll
  for (int rt = 0; rt < 2; ++rt)
#pragma unroll
    for (int ct = 0; ct < 4; ++ct)
#pragma unroll
      for (int r = 0; r < 8; ++r) acc[rt][ct][r] = 0.f;

  const int nk = K / 32;
  if (wid == 0) {
    tdm_load_2d_bf16(lds_off(&As[0][0][0]), A + (size_t)bM * K, 32, 128, K,
                     3, 3);
    tdm_load_2d_bf16(lds_off(&Bs[0][0][0]), W + bN, 128, 32, N, 5, 3);
  }

  for (int it = 0; it < nk; ++it) {
    const int cur = it & 1, nxt = cur ^ 1;
    if (wid == 0) {
      if (it + 1 < nk) {
        const int kb = (it + 1) * 32;
        tdm_load_2d_bf16(lds_off(&As[nxt][0][0]),
                         A + (size_t)bM * K + kb, 32, 128, K, 3, 3);
        tdm_load_2d_bf16(lds_off(&Bs[nxt][0][0]),
                         W + (size_t)kb * N + bN, 128, 32, N, 5, 3);
        __builtin_amdgcn_s_wait_tensorcnt((short)2);
      } else {
        __builtin_amdgcn_s_wait_tensorcnt((short)0);
      }
    }
    __syncthreads();

    AF a[2], b[4];
#pragma unroll
    for (int rt = 0; rt < 2; ++rt) {
      int row = wM + rt * 16 + l16;
#pragma unroll
      for (int p = 0; p < 8; ++p)
        a[rt].w[p] = *(const unsigned int*)&As[cur][row][akp(p, hi)];
    }
#pragma unroll
    for (int ct = 0; ct < 4; ++ct) {
      int n = wN + ct * 16 + l16;
#pragma unroll
      for (int i = 0; i < 16; ++i) b[ct].u[i] = Bs[cur][hi * 16 + i][n];
    }
#pragma unroll
    for (int rt = 0; rt < 2; ++rt)
#pragma unroll
      for (int ct = 0; ct < 4; ++ct)
        acc[rt][ct] = wmma_bf16(a[rt].v, b[ct].v, acc[rt][ct]);
    __syncthreads();
  }

// epilogue
#pragma unroll
  for (int rt = 0; rt < 2; ++rt) {
#pragma unroll
    for (int ct = 0; ct < 4; ++ct) {
      int gn = bN + wN + ct * 16 + l16;
      float bv = 0.f;
      if (BIAS) bv = bias[gn];
      int gm0 = bM + wM + rt * 16 + hi * 8;
#pragma unroll
      for (int r = 0; r < 8; ++r) {
        float val = acc[rt][ct][r] + bv;
        size_t idx = (size_t)(gm0 + r) * N + gn;
        if (RESID) val += resid[idx];
        if (GELU_) {
          float tt = 0.7978845608028654f * (val + 0.044715f * val * val * val);
          val = 0.5f * val * (1.f + tanhf(tt));
        }
        if (OUTBF) outB[idx] = f2bf(val);
        else outF[idx] = val;
      }
    }
  }
}

// ---------------- causal flash attention ----------------
// grid: (SEQ/128, BATCH*N_HEADS); 8 waves/block, 16 query rows/wave,
// K/V streamed via async-to-LDS (ASYNCcnt), double-buffered 32-key tiles.
__global__ __launch_bounds__(TB) void attn_flash(
    const unsigned short* __restrict__ Q, const unsigned short* __restrict__ Kg,
    const unsigned short* __restrict__ Vg, unsigned short* __restrict__ ctx) {
  __shared__ unsigned short Kt[2][32][72];
  __shared__ unsigned short Vt[2][32][72];
  __shared__ unsigned short Pt[8][16][36];

  const int tid = threadIdx.x;
  const int wid = tid >> 5, lane = tid & 31;
  const int hi = lane >> 4, l16 = lane & 15;
  const int b = blockIdx.y >> 4, h = blockIdx.y & 15;
  const int q0 = blockIdx.x * 128;
  const int qw0 = q0 + wid * 16;
  const size_t baseBS = (size_t)b * SEQ * D_EMB;

  // Q fragments for this wave's 16 rows (hd=64 -> two 16x32 A-frags)
  AF aQ[2];
  {
    const unsigned short* qrow =
        Q + baseBS + (size_t)(qw0 + l16) * D_EMB + h * HEAD_DIM;
#pragma unroll
    for (int kc = 0; kc < 2; ++kc)
#pragma unroll
      for (int p = 0; p < 8; ++p)
        aQ[kc].w[p] = *(const unsigned int*)(qrow + kc * 32 + akp(p, hi));
  }

  float m[8], l[8];
  v8f acc[4];
#pragma unroll
  for (int r = 0; r < 8; ++r) { m[r] = -1e30f; l[r] = 0.f; }
#pragma unroll
  for (int t = 0; t < 4; ++t)
#pragma unroll
    for (int r = 0; r < 8; ++r) acc[t][r] = 0.f;

  const int krow = tid >> 3, kc0 = (tid & 7) * 8;  // coop-load coords
  const int nj = q0 / 32 + 4;

  {  // prologue: stage tile 0 into buffer 0
    const size_t gidx = baseBS + (size_t)krow * D_EMB + h * HEAD_DIM + kc0;
    async_load_b128(lds_off(&Kt[0][krow][kc0]), Kg + gidx);
    async_load_b128(lds_off(&Vt[0][krow][kc0]), Vg + gidx);
  }

  for (int j = 0; j < nj; ++j) {
    const int cur = j & 1, nxt = cur ^ 1;
    if (j + 1 < nj) {
      const size_t gidx =
          baseBS + (size_t)((j + 1) * 32 + krow) * D_EMB + h * HEAD_DIM + kc0;
      async_load_b128(lds_off(&Kt[nxt][krow][kc0]), Kg + gidx);
      async_load_b128(lds_off(&Vt[nxt][krow][kc0]), Vg + gidx);
      asm volatile("s_wait_asynccnt 0x2" ::: "memory");
    } else {
      asm volatile("s_wait_asynccnt 0x0" ::: "memory");
    }
    __syncthreads();

    if (j * 32 <= qw0 + 15) {  // wave-uniform: EXEC stays all-ones for WMMA
      v8f s[2];
#pragma unroll
      for (int g = 0; g < 2; ++g) {
#pragma unroll
        for (int r = 0; r < 8; ++r) s[g][r] = 0.f;
#pragma unroll
        for (int kc = 0; kc < 2; ++kc) {
          AF bk;
          const unsigned int* src =
              (const unsigned int*)&Kt[cur][g * 16 + l16][kc * 32 + hi * 16];
#pragma unroll
          for (int p = 0; p < 8; ++p) bk.w[p] = src[p];
          s[g] = wmma_bf16(aQ[kc].v, bk.v, s[g]);
        }
      }
// scale (1/sqrt(64)) + causal mask
#pragma unroll
      for (int g = 0; g < 2; ++g)
#pragma unroll
        for (int r = 0; r < 8; ++r) {
          const int row = qw0 + r + hi * 8;
          const int col = j * 32 + g * 16 + l16;
          float sv = s[g][r] * 0.125f;
          s[g][r] = (col > row) ? -1e30f : sv;
        }
// online softmax per row
#pragma unroll
      for (int r = 0; r < 8; ++r) {
        float tmax = red16_max(fmaxf(s[0][r], s[1][r]));
        float nm = fmaxf(m[r], tmax);
        float alpha = __expf(m[r] - nm);
        float p0 = __expf(s[0][r] - nm);
        float p1 = __expf(s[1][r] - nm);
        l[r] = l[r] * alpha + red16_sum(p0 + p1);
        m[r] = nm;
#pragma unroll
        for (int t = 0; t < 4; ++t) acc[t][r] *= alpha;
        s[0][r] = p0;
        s[1][r] = p1;
      }
// P (D-layout) -> LDS -> A-fragment layout
#pragma unroll
      for (int g = 0; g < 2; ++g)
#pragma unroll
        for (int r = 0; r < 8; ++r)
          Pt[wid][r + hi * 8][g * 16 + l16] = f2bf(s[g][r]);
      asm volatile("s_wait_dscnt 0x0" ::: "memory");
      AF aP;
#pragma unroll
      for (int p = 0; p < 8; ++p)
        aP.w[p] = *(const unsigned int*)&Pt[wid][l16][akp(p, hi)];
// O += P * V
#pragma unroll
      for (int t = 0; t < 4; ++t) {
        AF bv;
#pragma unroll
        for (int i = 0; i < 16; ++i)
          bv.u[i] = Vt[cur][hi * 16 + i][t * 16 + l16];
        acc[t] = wmma_bf16(aP.v, bv.v, acc[t]);
      }
    }
    __syncthreads();
  }

// normalize + store ctx (bf16, [B,S,D] layout)
#pragma unroll
  for (int r = 0; r < 8; ++r) {
    const float inv = 1.0f / l[r];
    unsigned short* orow =
        ctx + baseBS + (size_t)(qw0 + r + hi * 8) * D_EMB + h * HEAD_DIM;
#pragma unroll
    for (int t = 0; t < 4; ++t) orow[t * 16 + l16] = f2bf(acc[t][r] * inv);
  }
}

// ---------------- host orchestration ----------------
extern "C" void kernel_launch(void* const* d_in, const int* in_sizes, int n_in,
                              void* d_out, int out_size, void* d_ws,
                              size_t ws_size, hipStream_t stream) {
  const float* x = (const float*)d_in[0];
  const float* wq = (const float*)d_in[1];
  const float* wk = (const float*)d_in[2];
  const float* wv = (const float*)d_in[3];
  const float* wo = (const float*)d_in[4];
  const float* bo = (const float*)d_in[5];
  const float* w1 = (const float*)d_in[6];
  const float* b1 = (const float*)d_in[7];
  const float* w2 = (const float*)d_in[8];
  const float* b2 = (const float*)d_in[9];
  const float* g1 = (const float*)d_in[10];
  const float* s1 = (const float*)d_in[11];
  const float* g2 = (const float*)d_in[12];
  const float* s2 = (const float*)d_in[13];

  const int M = BATCH * SEQ;             // 8192
  const size_t BSD = (size_t)M * D_EMB;  // 8M elems

  char* p = (char*)d_ws;
  auto alloc = [&](size_t bytes) -> void* {
    void* r = (void*)p;
    p += (bytes + 255) & ~(size_t)255;
    return r;
  };
  unsigned short* hb   = (unsigned short*)alloc(BSD * 2);
  unsigned short* qb   = (unsigned short*)alloc(BSD * 2);
  unsigned short* kb   = (unsigned short*)alloc(BSD * 2);
  unsigned short* vb   = (unsigned short*)alloc(BSD * 2);
  unsigned short* ctxb = (unsigned short*)alloc(BSD * 2);
  unsigned short* h2b  = (unsigned short*)alloc(BSD * 2);
  float* x2            = (float*)alloc(BSD * 4);
  unsigned short* ff1  = (unsigned short*)alloc((size_t)M * FFDIM * 2);
  unsigned short* wqb  = (unsigned short*)alloc((size_t)D_EMB * D_EMB * 2);
  unsigned short* wkb  = (unsigned short*)alloc((size_t)D_EMB * D_EMB * 2);
  unsigned short* wvb  = (unsigned short*)alloc((size_t)D_EMB * D_EMB * 2);
  unsigned short* wob  = (unsigned short*)alloc((size_t)D_EMB * D_EMB * 2);
  unsigned short* w1b  = (unsigned short*)alloc((size_t)D_EMB * FFDIM * 2);
  unsigned short* w2b  = (unsigned short*)alloc((size_t)FFDIM * D_EMB * 2);

  const int nDD = D_EMB * D_EMB, nDF = D_EMB * FFDIM;
  cvt_f32_bf16<<<256, TB, 0, stream>>>(wq, wqb, nDD);
  cvt_f32_bf16<<<256, TB, 0, stream>>>(wk, wkb, nDD);
  cvt_f32_bf16<<<256, TB, 0, stream>>>(wv, wvb, nDD);
  cvt_f32_bf16<<<256, TB, 0, stream>>>(wo, wob, nDD);
  cvt_f32_bf16<<<512, TB, 0, stream>>>(w1, w1b, nDF);
  cvt_f32_bf16<<<512, TB, 0, stream>>>(w2, w2b, nDF);

  layernorm_bf16<<<M, TB, 0, stream>>>(x, g1, s1, hb);

  dim3 gDD(D_EMB / 128, M / 128);
  gemm_wmma_bf16<false, false, false, true><<<gDD, TB, 0, stream>>>(
      hb, wqb, nullptr, nullptr, nullptr, qb, M, D_EMB, D_EMB);
  gemm_wmma_bf16<false, false, false, true><<<gDD, TB, 0, stream>>>(
      hb, wkb, nullptr, nullptr, nullptr, kb, M, D_EMB, D_EMB);
  gemm_wmma_bf16<false, false, false, true><<<gDD, TB, 0, stream>>>(
      hb, wvb, nullptr, nullptr, nullptr, vb, M, D_EMB, D_EMB);

  dim3 ga(SEQ / 128, BATCH * N_HEADS);
  attn_flash<<<ga, TB, 0, stream>>>(qb, kb, vb, ctxb);

  gemm_wmma_bf16<true, true, false, false><<<gDD, TB, 0, stream>>>(
      ctxb, wob, bo, x, x2, nullptr, M, D_EMB, D_EMB);

  layernorm_bf16<<<M, TB, 0, stream>>>(x2, g2, s2, h2b);

  dim3 gDF(FFDIM / 128, M / 128);
  gemm_wmma_bf16<true, false, true, true><<<gDF, TB, 0, stream>>>(
      h2b, w1b, b1, nullptr, nullptr, ff1, M, FFDIM, D_EMB);

  gemm_wmma_bf16<true, true, false, false><<<gDD, TB, 0, stream>>>(
      ff1, w2b, b2, x2, (float*)d_out, nullptr, M, D_EMB, FFDIM);

  (void)in_sizes; (void)n_in; (void)out_size; (void)ws_size;
}
